// BasicGraphModel_79680233276022
// MI455X (gfx1250) — matrix operand order repaired and checked by reading, not compiled, not run on hardware
//
#include <hip/hip_runtime.h>
#include <hip/hip_bf16.h>

#define N_NODES   100000
#define N_EDGES   3200000
#define IN_DIM    128
#define HID       32
#define N_CLASSES 8

typedef __attribute__((ext_vector_type(16))) _Float16 v16h;
typedef __attribute__((ext_vector_type(8)))  float    v8f;

// ---------------------------------------------------------------- utilities

__global__ void zero_f32_kernel(float* __restrict__ p, int n) {
    int i = blockIdx.x * blockDim.x + threadIdx.x;
    if (i < n) p[i] = 0.0f;
}

__global__ void zero_i32_kernel(int* __restrict__ p, int n) {
    int i = blockIdx.x * blockDim.x + threadIdx.x;
    if (i < n) p[i] = 0;
}

// out-degree as float (only needed for ninv_out); in-degree as int (needed for CSR)
__global__ void degree_kernel(const int* __restrict__ src, const int* __restrict__ dst,
                              float* __restrict__ fdeg_out, int* __restrict__ cnt_in) {
    int e = blockIdx.x * blockDim.x + threadIdx.x;
    if (e < N_EDGES) {
        atomicAdd(&fdeg_out[src[e]], 1.0f);
        atomicAdd(&cnt_in [dst[e]], 1);
    }
}

__global__ void ninv_kernel(float* __restrict__ ninv_out, float* __restrict__ ninv_in,
                            const int* __restrict__ cnt_in) {
    int i = blockIdx.x * blockDim.x + threadIdx.x;
    if (i < N_NODES) {
        ninv_out[i] = rsqrtf(fmaxf(ninv_out[i], 1.0f));          // in place over fdeg_out
        ninv_in[i]  = rsqrtf(fmaxf((float)cnt_in[i], 1.0f));
    }
}

// single-block exclusive scan of cnt_in[0..N_NODES) -> off[0..N_NODES], copy to cursor
#define SCAN_T 1024
__global__ __launch_bounds__(SCAN_T)
void scan_kernel(const int* __restrict__ cnt, int* __restrict__ off, int* __restrict__ cursor) {
    __shared__ int partial[SCAN_T];
    int t = threadIdx.x;
    const int chunk = (N_NODES + SCAN_T - 1) / SCAN_T;           // 98
    int lo = t * chunk;
    int hi = lo + chunk; if (hi > N_NODES) hi = N_NODES;
    int s = 0;
    for (int i = lo; i < hi; ++i) s += cnt[i];
    partial[t] = s;
    __syncthreads();
    for (int d = 1; d < SCAN_T; d <<= 1) {                       // Hillis-Steele inclusive
        int v = (t >= d) ? partial[t - d] : 0;
        __syncthreads();
        partial[t] += v;
        __syncthreads();
    }
    int run = (t == 0) ? 0 : partial[t - 1];
    for (int i = lo; i < hi; ++i) {
        off[i] = run; cursor[i] = run;
        run += cnt[i];
    }
    if (t == SCAN_T - 1) off[N_NODES] = partial[SCAN_T - 1];
}

__global__ void csr_fill_kernel(const int* __restrict__ src, const int* __restrict__ dst,
                                int* __restrict__ cursor, int* __restrict__ csr_src) {
    int e = blockIdx.x * blockDim.x + threadIdx.x;
    if (e < N_EDGES) {
        int p = atomicAdd(&cursor[dst[e]], 1);
        csr_src[p] = src[e];
    }
}

// ---------------------------------------------------------------- WMMA GEMM
// T[row, 0:32] = (X[row, :] * ninv_out[row]) @ W   for 16 rows per wave.
// A fragment: CDNA5 16-bit A 16x32 layout — lane L holds row M=L%16;
//   halves 0..7  -> K = kb + (L/16)*8 + e ; halves 8..15 -> K = kb + 16 + (L/16)*8 + e
// B fragment: lane L holds col N = nb + L%16; halves 0..7 -> K = kb + (L/16)*8 + e ; halves 8..15 -> K = kb + 16 + (L/16)*8 + e
// C/D: VGPR r, lane L -> row M = r + 8*(L/16), col N = L%16.

__device__ __forceinline__ v16h make_a_frag(const float* __restrict__ X, int ldx,
                                            int row0, int kb,
                                            const float* __restrict__ ninv) {
    int lane = threadIdx.x & 31;
    int m    = lane & 15;
    int hi   = lane >> 4;
    int row  = row0 + m;
    float s  = ninv[row];
    const float* p = X + (size_t)row * ldx + kb + hi * 8;
    v16h a;
#pragma unroll
    for (int e = 0; e < 8; ++e) a[e]     = (_Float16)(p[e]      * s);
#pragma unroll
    for (int e = 0; e < 8; ++e) a[8 + e] = (_Float16)(p[16 + e] * s);
    return a;
}

__device__ __forceinline__ v16h make_b_frag(const float* __restrict__ W, int ldw,
                                            int kb, int nb) {
    int lane = threadIdx.x & 31;
    const float* p = W + (size_t)(kb + (lane >> 4) * 8) * ldw + nb + (lane & 15);
    v16h b;
#pragma unroll
    for (int e = 0; e < 8; ++e) b[e]     = (_Float16)p[(size_t)e * ldw];
#pragma unroll
    for (int e = 0; e < 8; ++e) b[8 + e] = (_Float16)p[(size_t)(16 + e) * ldw];
    return b;
}

__device__ __forceinline__ void store_tile(float* __restrict__ T, int row0, int nb, v8f c) {
    int lane = threadIdx.x & 31;
    int n    = lane & 15;
    int mh   = (lane >> 4) * 8;
#pragma unroll
    for (int r = 0; r < 8; ++r)
        T[(size_t)(row0 + mh + r) * HID + nb + n] = c[r];
}

template <int K>
__global__ void gemm_wmma_kernel(const float* __restrict__ X,
                                 const float* __restrict__ W,
                                 const float* __restrict__ ninv,
                                 float* __restrict__ T) {
    int wave = blockIdx.x * (blockDim.x >> 5) + (threadIdx.x >> 5);
    int row0 = wave * 16;
    if (row0 >= N_NODES) return;          // wave-uniform: EXEC stays all-ones

    v8f c0 = {}, c1 = {};
#pragma unroll
    for (int kb = 0; kb < K; kb += 32) {
        v16h a  = make_a_frag(X, K, row0, kb, ninv);
        v16h b0 = make_b_frag(W, HID, kb, 0);
        v16h b1 = make_b_frag(W, HID, kb, 16);
        c0 = __builtin_amdgcn_wmma_f32_16x16x32_f16(false, a, false, b0, (short)0, c0, false, false);
        c1 = __builtin_amdgcn_wmma_f32_16x16x32_f16(false, a, false, b1, (short)0, c1, false, false);
    }
    store_tile(T, row0, 0,  c0);
    store_tile(T, row0, 16, c1);
}

// ---------------------------------------------------------------- aggregation (CSR, atomic-free)
// One wave per destination node; lane = feature. Register accumulation over
// in-edges; each T[src] row is a coalesced 128B load. Epilogue fused.

__global__ void agg_finalize_kernel(const int* __restrict__ off,
                                    const int* __restrict__ csr_src,
                                    const float* __restrict__ T,
                                    const float* __restrict__ ninv_in,
                                    const float* __restrict__ b,
                                    float* __restrict__ H, int do_relu) {
    int node = blockIdx.x * (blockDim.x >> 5) + (threadIdx.x >> 5);
    if (node >= N_NODES) return;
    int lane = threadIdx.x & 31;
    int beg = off[node];
    int end = off[node + 1];

    float acc = 0.0f;
    int i = beg;
    for (; i + 4 <= end; i += 4) {                 // unroll x4 for memory-level parallelism
        int s0 = csr_src[i + 0];
        int s1 = csr_src[i + 1];
        int s2 = csr_src[i + 2];
        int s3 = csr_src[i + 3];
        float v0 = T[(size_t)s0 * HID + lane];
        float v1 = T[(size_t)s1 * HID + lane];
        float v2 = T[(size_t)s2 * HID + lane];
        float v3 = T[(size_t)s3 * HID + lane];
        acc += v0 + v1 + v2 + v3;
    }
    for (; i < end; ++i)
        acc += T[(size_t)csr_src[i] * HID + lane];

    float v = fmaf(acc, ninv_in[node], b[lane]);
    if (do_relu) v = fmaxf(v, 0.0f);
    H[(size_t)node * HID + lane] = v;
}

// ---------------------------------------------------------------- pooling + head

__global__ void maxpool_kernel(const float* __restrict__ H, unsigned* __restrict__ G) {
    __shared__ float s[256];
    int tid = threadIdx.x;
    int c   = tid & 31;
    int row = tid >> 5;                     // 0..7
    float local = 0.0f;                     // relu fused: max with 0
    for (int n = blockIdx.x * 8 + row; n < N_NODES; n += gridDim.x * 8)
        local = fmaxf(local, H[(size_t)n * HID + c]);
    s[tid] = local;
    __syncthreads();
    for (int off = 128; off >= 32; off >>= 1) {
        if (tid < off) s[tid] = fmaxf(s[tid], s[tid + off]);
        __syncthreads();
    }
    if (tid < 32) atomicMax(&G[c], __float_as_uint(s[tid]));  // values >= 0: bit order == float order
}

__global__ void head_kernel(const unsigned* __restrict__ G,
                            const float* __restrict__ Wfc,
                            const float* __restrict__ bfc,
                            float* __restrict__ out) {
    __shared__ float g[HID];
    __shared__ float logit[N_CLASSES];
    int t = threadIdx.x;
    if (t < HID) g[t] = __uint_as_float(G[t]);
    __syncthreads();
    if (t < N_CLASSES) {
        float acc = bfc[t];
        for (int k = 0; k < HID; ++k) acc += g[k] * Wfc[k * N_CLASSES + t];
        logit[t] = acc;
    }
    __syncthreads();
    if (t < N_CLASSES) {
        float mx = logit[0];
        for (int i = 1; i < N_CLASSES; ++i) mx = fmaxf(mx, logit[i]);
        float ssum = 0.0f;
        for (int i = 0; i < N_CLASSES; ++i) ssum += __expf(logit[i] - mx);
        out[t] = __expf(logit[t] - mx) / ssum;
    }
}

// ---------------------------------------------------------------- launcher

extern "C" void kernel_launch(void* const* d_in, const int* in_sizes, int n_in,
                              void* d_out, int out_size, void* d_ws, size_t ws_size,
                              hipStream_t stream) {
    (void)in_sizes; (void)n_in; (void)out_size; (void)ws_size;

    const float* x   = (const float*)d_in[0];
    const int*   src = (const int*)  d_in[1];
    const int*   dst = (const int*)  d_in[2];
    const float* W0  = (const float*)d_in[3];
    const float* b0  = (const float*)d_in[4];
    const float* W1  = (const float*)d_in[5];
    const float* b1  = (const float*)d_in[6];
    const float* W2  = (const float*)d_in[7];
    const float* b2  = (const float*)d_in[8];
    const float* W3  = (const float*)d_in[9];
    const float* b3  = (const float*)d_in[10];
    const float* Wfc = (const float*)d_in[11];
    const float* bfc = (const float*)d_in[12];
    float* out = (float*)d_out;

    // workspace layout (dword offsets kept multiples of 4 -> 16B alignment)
    const size_t NH = (size_t)N_NODES * HID;          // 3,200,000
    float* ws       = (float*)d_ws;
    float* ninv_out = ws;                             // 100000 (deg accum -> rsqrt in place)
    float* ninv_in  = ninv_out + 100000;              // 100000
    int*   cnt_in   = (int*)(ninv_in + 100000);       // 100000
    int*   off      = cnt_in + 100000;                // 100004 (incl. sentinel + pad)
    int*   cursor   = off + 100004;                   // 100000
    int*   csr_src  = cursor + 100000;                // 3,200,000
    float* T        = (float*)(csr_src + N_EDGES);    // 3.2M (pre-aggregation h)
    float* HA       = T  + NH;                        // 3.2M
    float* HB       = HA + NH;                        // 3.2M
    unsigned* G     = (unsigned*)(HB + NH);           // 32

    const int blkNodes = (N_NODES + 255) / 256;       // 391
    const int blkEdges = (N_EDGES + 255) / 256;       // 12500
    const int tiles    = (N_NODES + 15) / 16;         // 6250
    const int blkGemm  = (tiles + 3) / 4;             // 4 waves / block
    const int blkAggN  = (N_NODES + 7) / 8;           // 8 waves / block -> 12500

    // ---- graph preprocessing: degrees, ninv, CSR by dst (amortized over 4 layers)
    zero_f32_kernel<<<blkNodes, 256, 0, stream>>>(ninv_out, N_NODES);
    zero_i32_kernel<<<blkNodes, 256, 0, stream>>>(cnt_in,   N_NODES);
    degree_kernel<<<blkEdges, 256, 0, stream>>>(src, dst, ninv_out, cnt_in);
    ninv_kernel<<<blkNodes, 256, 0, stream>>>(ninv_out, ninv_in, cnt_in);
    scan_kernel<<<1, SCAN_T, 0, stream>>>(cnt_in, off, cursor);
    csr_fill_kernel<<<blkEdges, 256, 0, stream>>>(src, dst, cursor, csr_src);

    // ---- layer 0: x(128) -> HA
    gemm_wmma_kernel<IN_DIM><<<blkGemm, 128, 0, stream>>>(x, W0, ninv_out, T);
    agg_finalize_kernel<<<blkAggN, 256, 0, stream>>>(off, csr_src, T, ninv_in, b0, HA, 1);

    // ---- layer 1: HA -> HB
    gemm_wmma_kernel<HID><<<blkGemm, 128, 0, stream>>>(HA, W1, ninv_out, T);
    agg_finalize_kernel<<<blkAggN, 256, 0, stream>>>(off, csr_src, T, ninv_in, b1, HB, 1);

    // ---- layer 2: HB -> HA
    gemm_wmma_kernel<HID><<<blkGemm, 128, 0, stream>>>(HB, W2, ninv_out, T);
    agg_finalize_kernel<<<blkAggN, 256, 0, stream>>>(off, csr_src, T, ninv_in, b2, HA, 1);

    // ---- layer 3: HA -> HB (no activation)
    gemm_wmma_kernel<HID><<<blkGemm, 128, 0, stream>>>(HA, W3, ninv_out, T);
    agg_finalize_kernel<<<blkAggN, 256, 0, stream>>>(off, csr_src, T, ninv_in, b3, HB, 0);

    // ---- relu + graph max pool + FC + softmax
    zero_f32_kernel<<<1, 32, 0, stream>>>((float*)G, HID);
    maxpool_kernel<<<256, 256, 0, stream>>>(HB, G);
    head_kernel<<<1, 32, 0, stream>>>(G, Wfc, bfc, out);
}
